// NormalGCN_79199196938457
// MI455X (gfx1250) — compile-verified, matrix operand order checked by reading, not run on hardware
//
#include <hip/hip_runtime.h>
#include <hip/hip_bf16.h>

#define N_NODES   50000
#define N_EDGES   800000
#define IN_DIM    64
#define HID       128
#define OUT_DIM   10
#define N_LAYERS  4
#define N_GRAPHS  256
#define BN_EPS    1e-5f
#define MT        5          // M-strips (of 16 rows) per GEMM block

static_assert(N_NODES % (16 * MT) == 0, "M must tile by 16*MT");

typedef __bf16 bf16_t;
typedef __attribute__((ext_vector_type(16))) __bf16 bf16x16;
typedef __attribute__((ext_vector_type(8)))  __bf16 bf16x8;
typedef __attribute__((ext_vector_type(8)))  float  f32x8;
typedef __attribute__((ext_vector_type(4)))  unsigned int u32x4;
typedef __attribute__((ext_vector_type(8)))  int i32x8;
typedef __attribute__((ext_vector_type(4)))  int i32x4;

union Frag { bf16x16 v; bf16x8 h[2]; };

#if __has_builtin(__builtin_amdgcn_tensor_load_to_lds)
#define USE_TDM 1
#else
#define USE_TDM 0
#endif

// ---------------------------------------------------------------- TDM helper
// DMA a [HID][K] bf16 weight matrix (row-major, N-major "W^T") from global
// into LDS with a 16B pad after every K-element row (bank-conflict-free
// ds_load_b128 afterwards). D# built per CDNA5 ISA §8.3/8.4.
#if USE_TDM
template <int K>
__device__ __forceinline__ void tdm_load_weights(bf16_t* lds_dst, const bf16_t* gsrc) {
    constexpr unsigned PAD_ITV = (K == 128) ? 5u : 4u;  // 64 / 32 DWORDs per row
    constexpr unsigned PAD_AMT = 3u;                    // pad 4 DWORDs (16B)
    const unsigned long long ga = (unsigned long long)(size_t)gsrc;
    u32x4 g0;
    g0[0] = 1u;                                         // count=1 (valid), user mode
    g0[1] = (unsigned)(size_t)lds_dst;                  // lds_addr (flat low32 = LDS offset)
    g0[2] = (unsigned)ga;                               // global_addr[31:0]
    g0[3] = (unsigned)((ga >> 32) & 0x01FFFFFFu) | (2u << 30);  // global_addr[56:32] | type=2
    i32x8 g1;
    g1[0] = (int)((1u << 16)                            // data_size = 1 -> 2 bytes
                | (1u << 20)                            // pad_enable
                | (PAD_ITV << 22) | (PAD_AMT << 25));
    g1[1] = (int)(((unsigned)K & 0xFFFFu) << 16);       // tensor_dim0 = K   (bits 79:48)
    g1[2] = (int)((unsigned)HID << 16);                 // tensor_dim1 = HID (bits 111:80)
    g1[3] = (int)(((unsigned)K & 0xFFFFu) << 16);       // tile_dim0 = K     (bits 127:112)
    g1[4] = (int)HID;                                   // tile_dim1 = HID, tile_dim2 = 0
    g1[5] = (int)K;                                     // tensor_dim0_stride = K
    g1[6] = 0;
    g1[7] = 0;
    const i32x4 gz = {0, 0, 0, 0};
#if __clang_major__ >= 23
    const i32x8 gz8 = {0, 0, 0, 0, 0, 0, 0, 0};
    __builtin_amdgcn_tensor_load_to_lds(g0, g1, gz, gz, gz8, 0);
#else
    __builtin_amdgcn_tensor_load_to_lds(g0, g1, gz, gz, 0);
#endif
}
#endif

// ---------------------------------------------------------------- utilities
__global__ void zero_f32_kernel(float* __restrict__ p, int n) {
    int i = blockIdx.x * 256 + threadIdx.x;
    if (i < n) p[i] = 0.0f;
}

__global__ void f32_to_bf16_kernel(const float* __restrict__ in,
                                   bf16_t* __restrict__ out, int n) {
    int i = blockIdx.x * 256 + threadIdx.x;
    if (i < n) out[i] = (bf16_t)in[i];
}

// in: [K][N] f32 (row-major, K = reduction dim)  ->  out: [N][K] bf16 (N-major)
__global__ void transpose_to_bf16_kernel(const float* __restrict__ in,
                                         bf16_t* __restrict__ out, int K, int N) {
    int t = blockIdx.x * 256 + threadIdx.x;
    if (t >= K * N) return;
    int k = t / N, n = t % N;
    out[(size_t)n * K + k] = (bf16_t)in[t];
}

__global__ void indeg_kernel(const int* __restrict__ dst, float* __restrict__ indeg) {
    int e = blockIdx.x * 256 + threadIdx.x;
    if (e < N_EDGES) atomicAdd(&indeg[dst[e]], 1.0f);
}

// ------------------------------------------------------------ WMMA GEMM core
// out = A[M][K](bf16) @ W (stored N-major [HID][K] bf16) + bias, f32 out.
// Block: 256 thr = 8 waves. Weights staged in LDS via TDM (tensor_load_to_lds,
// s_wait_tensorcnt), 16B row pad -> conflict-free ds_load_b128 B-fragments.
// Each block processes MT consecutive 16-row strips; wave w owns N-tile w.
// DUAL: two GEMMs (AW, BW) sharing the A fragments.
template <int K, bool DUAL>
__global__ __launch_bounds__(256)
void wmma_gemm_kernel(const bf16_t* __restrict__ A,
                      const bf16_t* __restrict__ WTa,   // [HID][K] bf16
                      const bf16_t* __restrict__ WTb,   // [HID][K] bf16 (DUAL)
                      const float*  __restrict__ bA,
                      const float*  __restrict__ bB,
                      float* __restrict__ outA,
                      float* __restrict__ outB) {
    constexpr int KP = K + 8;                 // padded LDS row stride (halfs)
    __shared__ bf16_t lds_wa[HID * KP];
    __shared__ bf16_t lds_wb[DUAL ? HID * KP : 8];

#if USE_TDM
    if (threadIdx.x < 32) {                   // wave 0 issues the DMA(s)
        tdm_load_weights<K>(lds_wa, WTa);
        if constexpr (DUAL) tdm_load_weights<K>(lds_wb, WTb);
        __builtin_amdgcn_s_wait_tensorcnt(0);
    }
#else
    for (int idx = threadIdx.x; idx < HID * (K / 8); idx += 256) {
        const int r = idx / (K / 8), c = (idx % (K / 8)) * 8;
        *(bf16x8*)(lds_wa + r * KP + c) = *(const bf16x8*)(WTa + (size_t)r * K + c);
        if constexpr (DUAL)
            *(bf16x8*)(lds_wb + r * KP + c) = *(const bf16x8*)(WTb + (size_t)r * K + c);
    }
#endif
    __syncthreads();

    const int lane = threadIdx.x & 31;
    const int wave = threadIdx.x >> 5;        // N-tile index 0..7
    const int nloc = lane & 15;
    const int ncol = wave * 16 + nloc;        // output column
    const int kb   = (lane >> 4) * 8;         // K sub-chunk per lane half
    const int hi   = (lane >> 4) * 8;

    const bf16_t* wra = lds_wa + (size_t)ncol * KP;
    const bf16_t* wrb = DUAL ? (lds_wb + (size_t)ncol * KP) : nullptr;
    const float biasA = bA[ncol];
    float biasB = 0.0f;
    if constexpr (DUAL) biasB = bB[ncol];

    for (int t = 0; t < MT; ++t) {
        const int m0 = (blockIdx.x * MT + t) * 16;
        const bf16_t* arow = A + (size_t)(m0 + nloc) * K;

        f32x8 accA = {};
        f32x8 accB = {};
#pragma unroll
        for (int k0 = 0; k0 < K; k0 += 32) {
            Frag af, bfA;
            // A frag: lane holds A[m][k0+kb .. +7] and A[m][k0+kb+16 .. +23]
            af.h[0]  = *(const bf16x8*)(arow + k0 + kb);
            af.h[1]  = *(const bf16x8*)(arow + k0 + kb + 16);
            // B frag: identical pattern on the N-major weight rows in LDS
            bfA.h[0] = *(const bf16x8*)(wra + k0 + kb);
            bfA.h[1] = *(const bf16x8*)(wra + k0 + kb + 16);
            accA = __builtin_amdgcn_wmma_f32_16x16x32_bf16(
                false, af.v, false, bfA.v, (short)0, accA, false, false);
            if constexpr (DUAL) {
                Frag bfB;
                bfB.h[0] = *(const bf16x8*)(wrb + k0 + kb);
                bfB.h[1] = *(const bf16x8*)(wrb + k0 + kb + 16);
                accB = __builtin_amdgcn_wmma_f32_16x16x32_bf16(
                    false, af.v, false, bfB.v, (short)0, accB, false, false);
            }
        }

        // C/D layout: acc[i] -> row m0 + i + (lane>=16 ? 8 : 0), col ncol
#pragma unroll
        for (int i = 0; i < 8; ++i) {
            const size_t idx = (size_t)(m0 + hi + i) * HID + ncol;
            outA[idx] = accA[i] + biasA;
            if constexpr (DUAL) outB[idx] = accB[i] + biasB;
        }
    }
}

// -------------------------------------------------- edge gather/scatter-add
// One wave per edge: 32 lanes x float4 = 128 floats/row. AX[dst] += BX[src].
__global__ __launch_bounds__(256)
void edge_scatter_kernel(const int* __restrict__ src, const int* __restrict__ dst,
                         const float* __restrict__ BX, float* __restrict__ AX) {
    const int lane = threadIdx.x & 31;
    const int e = blockIdx.x * 8 + (threadIdx.x >> 5);
    if (e >= N_EDGES) return;
    if (e + 8 < N_EDGES)  // hint next source row into cache (global_prefetch_b8)
        __builtin_prefetch(BX + (size_t)src[e + 8] * HID + lane * 4, 0, 1);
    const int s = src[e], d = dst[e];
    const float4 v = *(const float4*)(BX + (size_t)s * HID + lane * 4);
    float* dp = AX + (size_t)d * HID + lane * 4;
    atomicAdd(dp + 0, v.x);
    atomicAdd(dp + 1, v.y);
    atomicAdd(dp + 2, v.z);
    atomicAdd(dp + 3, v.w);
}

// --------------------------------- Hn = (has_msg ? AX+agg : H) * snorm + BN stats
#define NODES_PER_STAT_BLK 80
__global__ __launch_bounds__(HID)
void combine_stats_kernel(const float* __restrict__ AXagg, const float* __restrict__ H,
                          const float* __restrict__ indeg, const float* __restrict__ snorm,
                          float* __restrict__ Hn, float* __restrict__ bnstat) {
    const int c  = threadIdx.x;        // channel
    const int n0 = blockIdx.x * NODES_PER_STAT_BLK;
    float s = 0.0f, q = 0.0f;
    for (int i = 0; i < NODES_PER_STAT_BLK; ++i) {
        const int node = n0 + i;
        if (node >= N_NODES) break;
        const size_t idx = (size_t)node * HID + c;
        const float v = (indeg[node] > 0.0f ? AXagg[idx] : H[idx]) * snorm[node];
        Hn[idx] = v;
        s += v;
        q += v * v;
    }
    atomicAdd(&bnstat[c], s);
    atomicAdd(&bnstat[HID + c], q);
}

// ------------------------------------- H += relu(batchnorm(Hn) * gamma + beta)
__global__ void bn_apply_kernel(const float* __restrict__ Hn, const float* __restrict__ bnstat,
                                const float* __restrict__ gamma, const float* __restrict__ beta,
                                float* __restrict__ H) {
    const size_t idx = (size_t)blockIdx.x * 256 + threadIdx.x;
    if (idx >= (size_t)N_NODES * HID) return;
    const int c = (int)(idx & (HID - 1));
    const float inv_n = 1.0f / (float)N_NODES;
    const float mean  = bnstat[c] * inv_n;
    const float var   = bnstat[HID + c] * inv_n - mean * mean;
    const float v = (Hn[idx] - mean) * rsqrtf(var + BN_EPS) * gamma[c] + beta[c];
    H[idx] += fmaxf(v, 0.0f);
}

// ----------------------------------------------------------- graph pooling
__global__ void pool_count_kernel(const int* __restrict__ gid, float* __restrict__ counts) {
    int n = blockIdx.x * 256 + threadIdx.x;
    if (n < N_NODES) atomicAdd(&counts[gid[n]], 1.0f);
}

__global__ void pool_sum_kernel(const int* __restrict__ gid, const float* __restrict__ H,
                                float* __restrict__ pooled) {
    const size_t idx = (size_t)blockIdx.x * 256 + threadIdx.x;
    if (idx >= (size_t)N_NODES * HID) return;
    const int n = (int)(idx >> 7);
    const int c = (int)(idx & (HID - 1));
    atomicAdd(&pooled[(size_t)gid[n] * HID + c], H[idx]);
}

// ----------------------------------------------------------- readout MLP head
__global__ __launch_bounds__(HID)
void mlp_kernel(const float* __restrict__ pooled, const float* __restrict__ counts,
                const float* __restrict__ W0, const float* __restrict__ b0,
                const float* __restrict__ W1, const float* __restrict__ b1,
                const float* __restrict__ Wo, const float* __restrict__ bo,
                float* __restrict__ out) {
    __shared__ float p[HID];
    __shared__ float t[HID];
    const int g = blockIdx.x;
    const int c = threadIdx.x;
    const float cnt = fmaxf(counts[g], 1.0f);
    p[c] = pooled[(size_t)g * HID + c] / cnt;
    __syncthreads();
    float acc = b0[c];
    for (int k = 0; k < HID; ++k) acc += p[k] * W0[k * HID + c];
    t[c] = fmaxf(acc, 0.0f);
    __syncthreads();
    acc = b1[c];
    for (int k = 0; k < HID; ++k) acc += t[k] * W1[k * HID + c];
    __syncthreads();
    p[c] = fmaxf(acc, 0.0f);
    __syncthreads();
    if (c < OUT_DIM) {
        float o = bo[c];
        for (int k = 0; k < HID; ++k) o += p[k] * Wo[k * OUT_DIM + c];
        out[(size_t)g * OUT_DIM + c] = o;
    }
}

// ================================================================ launcher
extern "C" void kernel_launch(void* const* d_in, const int* in_sizes, int n_in,
                              void* d_out, int out_size, void* d_ws, size_t ws_size,
                              hipStream_t stream) {
    (void)in_sizes; (void)n_in; (void)out_size; (void)ws_size;

    const float* X       = (const float*)d_in[0];
    const float* snorm   = (const float*)d_in[2];
    const int*   src     = (const int*)d_in[3];
    const int*   dst     = (const int*)d_in[4];
    const int*   gid     = (const int*)d_in[5];
    const float* emb_h_W = (const float*)d_in[7];
    const float* emb_h_b = (const float*)d_in[8];
    const float* AW      = (const float*)d_in[11];
    const float* Ab      = (const float*)d_in[12];
    const float* BW      = (const float*)d_in[13];
    const float* Bb      = (const float*)d_in[14];
    const float* gamma   = (const float*)d_in[15];
    const float* beta    = (const float*)d_in[16];
    const float* W0      = (const float*)d_in[17];
    const float* b0      = (const float*)d_in[18];
    const float* W1      = (const float*)d_in[19];
    const float* b1      = (const float*)d_in[20];
    const float* Wo      = (const float*)d_in[21];
    const float* bo      = (const float*)d_in[22];
    float* out = (float*)d_out;

    // ---- carve workspace (256B-aligned chunks)
    char* ws = (char*)d_ws;
    size_t off = 0;
    auto carve = [&](size_t bytes) -> char* {
        char* p = ws + off;
        off += (bytes + 255) & ~(size_t)255;
        return p;
    };
    float*  H      = (float*)carve(sizeof(float) * N_NODES * HID);
    float*  AX     = (float*)carve(sizeof(float) * N_NODES * HID);  // AX, then += agg
    float*  BXb    = (float*)carve(sizeof(float) * N_NODES * HID);  // BX, then reused as Hn
    bf16_t* Hbf    = (bf16_t*)carve(sizeof(bf16_t) * N_NODES * HID);
    bf16_t* Xbf    = (bf16_t*)carve(sizeof(bf16_t) * N_NODES * IN_DIM);
    bf16_t* eWT    = (bf16_t*)carve(sizeof(bf16_t) * HID * IN_DIM);
    bf16_t* AWT    = (bf16_t*)carve(sizeof(bf16_t) * N_LAYERS * HID * HID);
    bf16_t* BWT    = (bf16_t*)carve(sizeof(bf16_t) * N_LAYERS * HID * HID);
    float*  indeg  = (float*)carve(sizeof(float) * N_NODES);
    float*  bnstat = (float*)carve(sizeof(float) * 2 * HID);
    float*  pooled = (float*)carve(sizeof(float) * N_GRAPHS * HID);
    float*  counts = (float*)carve(sizeof(float) * N_GRAPHS);

    const int NE_BLK   = (N_EDGES + 255) / 256;       // 3125
    const int NN_BLK   = (N_NODES + 255) / 256;       // 196
    const int NH_BLK   = (N_NODES * HID) / 256;       // 25000 (exact)
    const int GEMM_BLK = N_NODES / 16 / MT;           // 625

    // ---- init
    zero_f32_kernel<<<NN_BLK, 256, 0, stream>>>(indeg, N_NODES);
    zero_f32_kernel<<<(N_GRAPHS * HID) / 256, 256, 0, stream>>>(pooled, N_GRAPHS * HID);
    zero_f32_kernel<<<1, 256, 0, stream>>>(counts, N_GRAPHS);

    // ---- weight prep: transpose to N-major + convert to bf16
    transpose_to_bf16_kernel<<<(IN_DIM * HID) / 256, 256, 0, stream>>>(emb_h_W, eWT, IN_DIM, HID);
    for (int l = 0; l < N_LAYERS; ++l) {
        transpose_to_bf16_kernel<<<(HID * HID) / 256, 256, 0, stream>>>(
            AW + (size_t)l * HID * HID, AWT + (size_t)l * HID * HID, HID, HID);
        transpose_to_bf16_kernel<<<(HID * HID) / 256, 256, 0, stream>>>(
            BW + (size_t)l * HID * HID, BWT + (size_t)l * HID * HID, HID, HID);
    }
    f32_to_bf16_kernel<<<(N_NODES * IN_DIM) / 256, 256, 0, stream>>>(X, Xbf, N_NODES * IN_DIM);
    indeg_kernel<<<NE_BLK, 256, 0, stream>>>(dst, indeg);

    // ---- embedding: H = X @ emb_h_W + b   (WMMA, K=64, weights via TDM->LDS)
    wmma_gemm_kernel<IN_DIM, false><<<GEMM_BLK, 256, 0, stream>>>(
        Xbf, eWT, nullptr, emb_h_b, nullptr, H, nullptr);

    // ---- GCN layers
    for (int l = 0; l < N_LAYERS; ++l) {
        f32_to_bf16_kernel<<<NH_BLK, 256, 0, stream>>>(H, Hbf, N_NODES * HID);
        zero_f32_kernel<<<1, 256, 0, stream>>>(bnstat, 2 * HID);
        // AX = H@AW + Ab ; BX = H@BW + Bb  (fused dual WMMA GEMM)
        wmma_gemm_kernel<HID, true><<<GEMM_BLK, 256, 0, stream>>>(
            Hbf, AWT + (size_t)l * HID * HID, BWT + (size_t)l * HID * HID,
            Ab + (size_t)l * HID, Bb + (size_t)l * HID, AX, BXb);
        // AX[dst] += BX[src] over all edges
        edge_scatter_kernel<<<N_EDGES / 8, 256, 0, stream>>>(src, dst, BXb, AX);
        // Hn = (has_msg ? AX : H) * snorm ; accumulate BN sum/sumsq
        combine_stats_kernel<<<N_NODES / NODES_PER_STAT_BLK, HID, 0, stream>>>(
            AX, H, indeg, snorm, BXb, bnstat);
        // H += relu(BN(Hn))
        bn_apply_kernel<<<NH_BLK, 256, 0, stream>>>(
            BXb, bnstat, gamma + (size_t)l * HID, beta + (size_t)l * HID, H);
    }

    // ---- mean-pool per graph + MLP head
    pool_count_kernel<<<NN_BLK, 256, 0, stream>>>(gid, counts);
    pool_sum_kernel<<<NH_BLK, 256, 0, stream>>>(gid, H, pooled);
    mlp_kernel<<<N_GRAPHS, HID, 0, stream>>>(pooled, counts, W0, b0, W1, b1, Wo, bo, out);
}
